// GroupedQueryAttention_43568148250652
// MI455X (gfx1250) — matrix operand order repaired and checked by reading, not compiled, not run on hardware
//
#include <hip/hip_runtime.h>
#include <hip/hip_bf16.h>

typedef __attribute__((ext_vector_type(16))) _Float16 v16h;
typedef __attribute__((ext_vector_type(8)))  _Float16 v8h;
typedef __attribute__((ext_vector_type(8)))  float    v8f;

#define KB   2
#define KS   4096
#define KE   640
#define KH   4
#define KD   256
#define KWIN 512
#define NTOK (KB*KS)          // 8192 tokens
#define SCALING 0.0625f       // 256^-0.5
#define KEPS 1e-6f

#define WMMA_F16(a,b,c) __builtin_amdgcn_wmma_f32_16x16x32_f16(false,(a),false,(b),(short)0,(c),false,false)

// A-operand load: lanes 0-15 want K {0..7,16..23}, lanes 16-31 K {8..15,24..31}.
// Caller passes p = row_base + hs*8; we fetch p[0..7] and p[16..23].
__device__ __forceinline__ v16h ldA(const _Float16* p) {
  union { v16h v; v8h h[2]; } u;
  u.h[0] = *(const v8h*)(p);
  u.h[1] = *(const v8h*)(p + 16);
  return u.v;
}
// B-operand load: same K order as ldA (caller passes row_base + hs*8).
__device__ __forceinline__ v16h ldB(const _Float16* p) { return ldA(p); }

// ---------------- elementwise conversion ----------------
__global__ void __launch_bounds__(256) cvt_f32_f16(const float* __restrict__ in,
                                                   _Float16* __restrict__ out, int n4) {
  int i = blockIdx.x * blockDim.x + threadIdx.x;
  if (i >= n4) return;
  float4 v = ((const float4*)in)[i];
  _Float16* op = out + (size_t)i * 4;
  op[0] = (_Float16)v.x; op[1] = (_Float16)v.y;
  op[2] = (_Float16)v.z; op[3] = (_Float16)v.w;
}

// ---------------- transpose fp32 [R][C] -> f16 [C][R] ----------------
__global__ void __launch_bounds__(256) transpose_cvt(const float* __restrict__ in,
                                                     _Float16* __restrict__ out, int R, int C) {
  __shared__ float t[16][17];
  int r0 = blockIdx.y * 16, c0 = blockIdx.x * 16;
  t[threadIdx.y][threadIdx.x] = in[(size_t)(r0 + threadIdx.y) * C + c0 + threadIdx.x];
  __syncthreads();
  out[(size_t)(c0 + threadIdx.y) * R + r0 + threadIdx.x] = (_Float16)t[threadIdx.x][threadIdx.y];
}

// ---------------- f16 batch-local transpose: [b][s][256] -> [b][256][s] ----------------
__global__ void __launch_bounds__(256) transpose_v(const _Float16* __restrict__ in,
                                                   _Float16* __restrict__ out) {
  int b = blockIdx.z;
  in  += (size_t)b * KS * KD;
  out += (size_t)b * KD * KS;
  __shared__ _Float16 t[16][17];
  int s0 = blockIdx.y * 16, d0 = blockIdx.x * 16;
  t[threadIdx.y][threadIdx.x] = in[(size_t)(s0 + threadIdx.y) * KD + d0 + threadIdx.x];
  __syncthreads();
  out[(size_t)(d0 + threadIdx.y) * KS + s0 + threadIdx.x] = t[threadIdx.x][threadIdx.y];
}

// ---------------- generic WMMA GEMM: C[M][N] = A[M][K] * B^T (B given as [N][K]) ------------
template<bool F16OUT>
__global__ void __launch_bounds__(128) gemm_wmma(const _Float16* __restrict__ A,
                                                 const _Float16* __restrict__ Bm,
                                                 void* __restrict__ Cm, int M, int N, int K) {
  const int lane = threadIdx.x & 31;
  const int wid  = threadIdx.x >> 5;
  const int wave = blockIdx.x * (blockDim.x >> 5) + wid;
  const int tn4 = N >> 6;          // groups of 64 output columns
  const int tm  = M >> 4;
  if (wave >= tm * tn4) return;
  const int m0 = (wave / tn4) << 4;
  const int n0 = (wave % tn4) << 6;
  const int l16 = lane & 15, hs = lane >> 4;

  v8f acc[4];
  #pragma unroll
  for (int t = 0; t < 4; t++)
    #pragma unroll
    for (int r = 0; r < 8; r++) acc[t][r] = 0.f;

  const _Float16* ap = A  + (size_t)(m0 + l16) * K + hs * 8;
  const _Float16* bp = Bm + (size_t)(n0 + l16) * K + hs * 8;
  const int nk = K >> 5;
  for (int kc = 0; kc < nk; ++kc) {
    v16h a = ldA(ap + kc * 32);
    #pragma unroll
    for (int t = 0; t < 4; t++) {
      v16h bb = ldB(bp + (size_t)(t * 16) * K + kc * 32);
      acc[t] = WMMA_F16(a, bb, acc[t]);
    }
  }
  #pragma unroll
  for (int t = 0; t < 4; t++)
    #pragma unroll
    for (int r = 0; r < 8; r++) {
      int m = m0 + r + (hs << 3);
      int n = n0 + t * 16 + l16;
      if (F16OUT) ((_Float16*)Cm)[(size_t)m * N + n] = (_Float16)acc[t][r];
      else        ((float*)Cm)[(size_t)m * N + n]    = acc[t][r];
    }
}

// ---------------- RMSNorm + RoPE (in-place on f16 rows of 256), fp32 math ----------------
__global__ void __launch_bounds__(256) norm_rope(_Float16* __restrict__ data,
                                                 const float* __restrict__ scale,
                                                 const float* __restrict__ cosp,
                                                 const float* __restrict__ sinp,
                                                 int heads, float outScale) {
  int row = blockIdx.x;                // NTOK*heads rows of 256
  int d = threadIdx.x;
  int s = (row / heads) % KS;          // sequence position
  _Float16* p = data + (size_t)row * KD;
  float x = (float)p[d];

  __shared__ float red[8];
  float ss = x * x;
  #pragma unroll
  for (int off = 16; off >= 1; off >>= 1) ss += __shfl_xor(ss, off, 32);
  if ((threadIdx.x & 31) == 0) red[threadIdx.x >> 5] = ss;
  __syncthreads();
  float tot = 0.f;
  #pragma unroll
  for (int i = 0; i < 8; i++) tot += red[i];
  float inv = rsqrtf(tot * (1.0f / KD) + KEPS);
  float xn = x * inv * (1.f + scale[d]);

  __shared__ float sh[KD];
  sh[d] = xn;
  __syncthreads();
  float other = sh[(d + 128) & 255];
  float rot = (d < 128) ? -other : other;
  float out = (xn * cosp[(size_t)s * KD + d] + rot * sinp[(size_t)s * KD + d]) * outScale;
  p[d] = (_Float16)out;
}

// ---------------- flash attention, sliding window 512, one wave per 16-query tile -----------
__global__ void __launch_bounds__(128, 1)
attn_kernel(const _Float16* __restrict__ qh, const _Float16* __restrict__ kh,
            const _Float16* __restrict__ vT, _Float16* __restrict__ aoh) {
  const int lane = threadIdx.x & 31;
  const int wid  = threadIdx.x >> 5;
  const int l16 = lane & 15, hs = lane >> 4;
  const int tile = blockIdx.x * 4 + wid;      // 2048 tiles: b*1024 + h*256 + qt
  const int b  = tile >> 10;
  const int h  = (tile >> 8) & 3;
  const int qt = tile & 255;
  const int qbase = qt << 4;

  __shared__ _Float16 qs[4][16 * KD];
  // stage this wave's 16x256 q tile into LDS (one row per iteration, coalesced b128)
  for (int t = 0; t < 16; t++) {
    const _Float16* src = qh + (((size_t)(b * KS + qbase + t)) * KH + h) * KD + lane * 8;
    *(v8h*)(qs[wid] + t * KD + lane * 8) = *(const v8h*)src;
  }
  __syncthreads();

  v8f o[16];
  #pragma unroll
  for (int t = 0; t < 16; t++)
    #pragma unroll
    for (int r = 0; r < 8; r++) o[t][r] = 0.f;
  float mrow = -1e30f, lrow = 0.f;

  int lo = qbase - (KWIN - 1); if (lo < 0) lo = 0;
  const int kb0 = lo & ~31;
  const int i_abs = qbase + l16;

  const _Float16* qp   = qs[wid] + l16 * KD + hs * 8;                  // B-operand: col=query
  const _Float16* kbse = kh + (size_t)(b * KS) * KD;                   // A-operand rows: keys
  const _Float16* vbse = vT + (size_t)b * KD * KS;

  for (int kb = kb0; kb <= qbase + 15; kb += 32) {
    // Compiler barrier: forbid hoisting the (loop-invariant) LDS q-chunk loads out of
    // this loop into VGPRs — that blows the register budget and round-trips q through
    // scratch. Re-reading LDS every iteration is cheap; scratch spills are not.
    asm volatile("" ::: "memory");

    // prefetch next key tile rows into cache while we compute this one
    if (kb + 32 <= qbase + 15) {
      __builtin_prefetch(kbse + (size_t)(kb + 32 + l16) * KD, 0, 0);
      __builtin_prefetch(kbse + (size_t)(kb + 48 + l16) * KD, 0, 0);
    }

    // ---- S^T = K * Q^T for 32 keys x 16 queries ----
    v8f st0, st1;
    #pragma unroll
    for (int r = 0; r < 8; r++) { st0[r] = 0.f; st1[r] = 0.f; }
    const _Float16* kp0 = kbse + (size_t)(kb + l16) * KD + hs * 8;
    const _Float16* kp1 = kbse + (size_t)(kb + 16 + l16) * KD + hs * 8;
    #pragma unroll
    for (int kc = 0; kc < 8; kc++) {
      v16h qb  = ldB(qp  + kc * 32);
      v16h ka0 = ldA(kp0 + kc * 32);
      v16h ka1 = ldA(kp1 + kc * 32);
      st0 = WMMA_F16(ka0, qb, st0);
      st1 = WMMA_F16(ka1, qb, st1);
    }
    // ---- mask (causal + window) ----
    #pragma unroll
    for (int r = 0; r < 8; r++) {
      int j0 = kb + r + (hs << 3);
      int j1 = j0 + 16;
      if (j0 > i_abs || (i_abs - j0) >= KWIN) st0[r] = -1e30f;
      if (j1 > i_abs || (i_abs - j1) >= KWIN) st1[r] = -1e30f;
    }
    // ---- online softmax: each lane owns 16 of 32 scores for query l16 ----
    float bm = -1e30f;
    #pragma unroll
    for (int r = 0; r < 8; r++) { bm = fmaxf(bm, st0[r]); bm = fmaxf(bm, st1[r]); }
    bm = fmaxf(bm, __shfl_xor(bm, 16, 32));
    float newm = fmaxf(mrow, bm);
    float alpha = __expf(mrow - newm);
    float psum = 0.f;
    v16h pa;                     // P already lands in the PV A-operand layout
    #pragma unroll
    for (int r = 0; r < 8; r++) {
      float p0 = __expf(st0[r] - newm);
      float p1 = __expf(st1[r] - newm);
      psum += p0 + p1;
      pa[r]     = (_Float16)p0;
      pa[8 + r] = (_Float16)p1;
    }
    psum += __shfl_xor(psum, 16, 32);
    lrow = lrow * alpha + psum;
    mrow = newm;
    // broadcast per-row alpha to C-layout lanes
    float av[8];
    #pragma unroll
    for (int r = 0; r < 8; r++) av[r] = __shfl(alpha, r + (hs << 3), 32);
    // ---- O = O*alpha + P*V over 16 d-tiles ----
    #pragma unroll
    for (int t = 0; t < 16; t++) {
      v16h vb = ldB(vbse + (size_t)(t * 16 + l16) * KS + kb + hs * 8);
      #pragma unroll
      for (int r = 0; r < 8; r++) o[t][r] *= av[r];
      o[t] = WMMA_F16(pa, vb, o[t]);
    }
  }

  float linv = 1.0f / lrow;
  float lv[8];
  #pragma unroll
  for (int r = 0; r < 8; r++) lv[r] = __shfl(linv, r + (hs << 3), 32);
  #pragma unroll
  for (int t = 0; t < 16; t++)
    #pragma unroll
    for (int r = 0; r < 8; r++) {
      int tok = qbase + r + (hs << 3);
      int col = h * KD + t * 16 + l16;
      aoh[(size_t)(b * KS + tok) * (KH * KD) + col] = (_Float16)(o[t][r] * lv[r]);
    }
}

// ---------------------------------------------------------------------------------------------
extern "C" void kernel_launch(void* const* d_in, const int* in_sizes, int n_in,
                              void* d_out, int out_size, void* d_ws, size_t ws_size,
                              hipStream_t stream) {
  (void)in_sizes; (void)n_in; (void)out_size; (void)ws_size;
  const float* x      = (const float*)d_in[0];
  const float* cosp   = (const float*)d_in[2];
  const float* sinp   = (const float*)d_in[3];
  const float* Wq     = (const float*)d_in[4];
  const float* Wk     = (const float*)d_in[5];
  const float* Wv     = (const float*)d_in[6];
  const float* Wo     = (const float*)d_in[7];
  const float* qscale = (const float*)d_in[8];
  const float* kscale = (const float*)d_in[9];
  float* out = (float*)d_out;

  char* ws = (char*)d_ws;
  size_t off = 0;
  auto take = [&](size_t bytes) -> char* {
    char* p = ws + off;
    off += (bytes + 255) & ~(size_t)255;
    return p;
  };
  _Float16* xh  = (_Float16*)take((size_t)NTOK * KE * 2);
  _Float16* WqT = (_Float16*)take((size_t)1024 * 640 * 2);
  _Float16* WkT = (_Float16*)take((size_t)256 * 640 * 2);
  _Float16* WvT = (_Float16*)take((size_t)256 * 640 * 2);
  _Float16* WoT = (_Float16*)take((size_t)640 * 1024 * 2);
  _Float16* qh  = (_Float16*)take((size_t)NTOK * 1024 * 2);
  _Float16* kh  = (_Float16*)take((size_t)NTOK * 256 * 2);
  _Float16* vh  = (_Float16*)take((size_t)NTOK * 256 * 2);
  _Float16* vTh = (_Float16*)take((size_t)NTOK * 256 * 2);
  _Float16* aoh = (_Float16*)take((size_t)NTOK * 1024 * 2);

  // 1) x -> f16
  {
    int n4 = NTOK * KE / 4;
    cvt_f32_f16<<<(n4 + 255) / 256, 256, 0, stream>>>(x, xh, n4);
  }
  // 2) weights -> transposed f16 [N][K]
  transpose_cvt<<<dim3(1024 / 16, 640 / 16), dim3(16, 16), 0, stream>>>(Wq, WqT, 640, 1024);
  transpose_cvt<<<dim3(256 / 16, 640 / 16),  dim3(16, 16), 0, stream>>>(Wk, WkT, 640, 256);
  transpose_cvt<<<dim3(256 / 16, 640 / 16),  dim3(16, 16), 0, stream>>>(Wv, WvT, 640, 256);
  transpose_cvt<<<dim3(640 / 16, 1024 / 16), dim3(16, 16), 0, stream>>>(Wo, WoT, 1024, 640);
  // 3) QKV projections (WMMA, f16 out)
  gemm_wmma<true><<<2048, 128, 0, stream>>>(xh, WqT, qh, NTOK, 1024, KE);
  gemm_wmma<true><<<512,  128, 0, stream>>>(xh, WkT, kh, NTOK, 256, KE);
  gemm_wmma<true><<<512,  128, 0, stream>>>(xh, WvT, vh, NTOK, 256, KE);
  // 4) RMSNorm + RoPE (SCALING folded into q)
  norm_rope<<<NTOK * KH, 256, 0, stream>>>(qh, qscale, cosp, sinp, KH, SCALING);
  norm_rope<<<NTOK,      256, 0, stream>>>(kh, kscale, cosp, sinp, 1, 1.0f);
  // 5) V -> [b][d][s]
  transpose_v<<<dim3(KD / 16, KS / 16, KB), dim3(16, 16), 0, stream>>>(vh, vTh);
  // 6) sliding-window flash attention
  attn_kernel<<<512, 128, 0, stream>>>(qh, kh, vTh, aoh);
  // 7) output projection (WMMA, fp32 out)
  gemm_wmma<false><<<1280, 128, 0, stream>>>(aoh, WoT, out, NTOK, KE, 1024);
}
